// MSDeformAttnTransformerEncoderOnly_77129022701636
// MI455X (gfx1250) — matrix operand order repaired and checked by reading, not compiled, not run on hardware
//
#include <hip/hip_runtime.h>
#include <hip/hip_bf16.h>
#include <cstdint>
#include <cstddef>

// ---------------- problem constants ----------------
#define B_N 2
#define E_N 256
#define NH_N 8
#define NL_N 4
#define NP_N 4
#define HD_N 32
#define DF_N 1024
#define NLAYERS_N 6
#define S_N 10548
#define M_N (B_N * S_N)   // 21096 rows
#define MPAD_N 21120      // 330 * 64 (GEMM tile padded)

static_assert(MPAD_N % 64 == 0 && MPAD_N >= M_N, "pad");

typedef __bf16 v16bf __attribute__((ext_vector_type(16)));
typedef float  v8f   __attribute__((ext_vector_type(8)));
typedef unsigned int v4u __attribute__((ext_vector_type(4)));
typedef int v4i __attribute__((ext_vector_type(4)));
typedef int v8i __attribute__((ext_vector_type(8)));

// Tensor Data Mover availability (this toolchain: 6-arg builtin)
#if defined(__has_builtin)
#if __has_builtin(__builtin_amdgcn_tensor_load_to_lds) && \
    __has_builtin(__builtin_amdgcn_s_wait_tensorcnt)
#define USE_TDM 1
#endif
#endif
#ifndef USE_TDM
#define USE_TDM 0
#endif

__device__ __constant__ int c_lvl_start[4] = {0, 9216, 10368, 10512};
__device__ __constant__ int c_lvl_D[4]     = {4, 2, 1, 1};
__device__ __constant__ int c_lvl_H[4]     = {48, 24, 12, 6};
__device__ __constant__ int c_lvl_W[4]     = {48, 24, 12, 6};
__device__ __constant__ int c_lvl_sz[4]    = {9216, 1152, 144, 36};

__device__ __forceinline__ unsigned short f2bf(float f) {
  unsigned u = __float_as_uint(f);
  u += 0x7FFFu + ((u >> 16) & 1u);   // round-to-nearest-even
  return (unsigned short)(u >> 16);
}

// ---------------- pack feats/pos -> (B,S,E) row-major, zero pad rows ----------
__global__ __launch_bounds__(256) void pack_x_pos(
    const float* __restrict__ f0, const float* __restrict__ f1,
    const float* __restrict__ f2, const float* __restrict__ f3,
    const float* __restrict__ p0, const float* __restrict__ p1,
    const float* __restrict__ p2, const float* __restrict__ p3,
    const float* __restrict__ lvl_emb, float* __restrict__ x,
    float* __restrict__ pos) {
  size_t idx = (size_t)blockIdx.x * 256 + threadIdx.x;
  if (idx >= (size_t)MPAD_N * E_N) return;
  int row = (int)(idx / E_N);
  int e   = (int)(idx - (size_t)row * E_N);
  if (row >= M_N) { x[idx] = 0.f; pos[idx] = 0.f; return; }
  int b = row / S_N;
  int s = row - b * S_N;
  int lvl = (s >= c_lvl_start[3]) ? 3 : (s >= c_lvl_start[2]) ? 2
          : (s >= c_lvl_start[1]) ? 1 : 0;
  int sp = s - c_lvl_start[lvl];
  const float* f = lvl == 0 ? f0 : lvl == 1 ? f1 : lvl == 2 ? f2 : f3;
  const float* p = lvl == 0 ? p0 : lvl == 1 ? p1 : lvl == 2 ? p2 : p3;
  size_t src = ((size_t)b * E_N + e) * c_lvl_sz[lvl] + sp;
  x[idx]   = f[src];
  pos[idx] = p[src] + lvl_emb[lvl * E_N + e];
}

// ------------- weight [K][N] f32 -> [N][K] bf16 (one-time) -------------------
__global__ __launch_bounds__(256) void wt_transpose(
    const float* __restrict__ W, unsigned short* __restrict__ Wt, int K, int N) {
  int idx = blockIdx.x * 256 + threadIdx.x;
  if (idx >= N * K) return;
  int n = idx / K, k = idx - n * K;
  Wt[idx] = f2bf(W[(size_t)k * N + n]);
}

// ------------- f32 -> bf16 converts (zero the pad rows) ----------------------
__global__ __launch_bounds__(256) void cvt_bf16_rows(
    const float* __restrict__ src, unsigned short* __restrict__ dst, int ncols) {
  size_t idx = (size_t)blockIdx.x * 256 + threadIdx.x;
  if (idx >= (size_t)MPAD_N * ncols) return;
  int row = (int)(idx / (unsigned)ncols);
  dst[idx] = f2bf(row < M_N ? src[idx] : 0.f);
}

__global__ __launch_bounds__(256) void add_cvt_bf16_rows(
    const float* __restrict__ a, const float* __restrict__ b,
    unsigned short* __restrict__ dst) {
  size_t idx = (size_t)blockIdx.x * 256 + threadIdx.x;
  if (idx >= (size_t)MPAD_N * E_N) return;
  int row = (int)(idx / E_N);
  dst[idx] = f2bf(row < M_N ? a[idx] + b[idx] : 0.f);
}

#if USE_TDM
// Issue one TDM 2D tile load: 64 rows x 32 bf16 from row-major [rows][kdim]
// into LDS at lds_off with 16B padding after every 64B row (-> 80B LDS stride).
__device__ __forceinline__ void tdm_tile_load(const unsigned short* gptr,
                                              unsigned lds_off, int kdim) {
  const unsigned long long ga = (unsigned long long)gptr;
  v4u g0;
  g0[0] = 1u;                                   // count=1, user descriptor
  g0[1] = lds_off;                              // lds_addr (bytes)
  g0[2] = (unsigned)(ga & 0xFFFFFFFFu);         // global_addr[31:0]
  g0[3] = (unsigned)((ga >> 32) & 0x1FFFFFFu)   // global_addr[56:32]
        | (2u << 30);                           // type = 2 ("image")
  v8i g1;
  // data_size=1 (2B) | pad_enable | pad_interval=3 (16 DW) | pad_amount=3 (4 DW)
  g1[0] = (1 << 16) | (1 << 20) | (3 << 22) | (3 << 25);
  g1[1] = (kdim & 0xFFFF) << 16;                // tensor_dim0[15:0] (elems)
  g1[2] = ((kdim >> 16) & 0xFFFF) | (64 << 16); // tensor_dim0[31:16] | tensor_dim1=64
  g1[3] = (32 << 16);                           // tensor_dim1 hi=0 | tile_dim0=32
  g1[4] = 64;                                   // tile_dim1=64 | tile_dim2=0
  g1[5] = kdim;                                 // tensor_dim0_stride[31:0] (elems)
  g1[6] = 0;                                    // stride hi | dim1_stride lo
  g1[7] = 0;
  v4i z4 = {0, 0, 0, 0};
  v8i z8 = {0, 0, 0, 0, 0, 0, 0, 0};
  __builtin_amdgcn_tensor_load_to_lds(g0, g1, z4, z4, z8, 0);
}
#endif

// ---------------- WMMA bf16 GEMM: C[M][N] = A[M][K] * Bt[N][K]^T + bias ------
// block: 128 threads (4 waves), tile 64x64, each wave 32x32 (2x2 WMMA tiles)
template <int KDIM, bool RELU, bool F32OUT, bool BF16OUT>
__global__ __launch_bounds__(128) void gemm_bf16_wmma(
    const unsigned short* __restrict__ A,   // [MPAD][KDIM] bf16 bits
    const unsigned short* __restrict__ Bt,  // [N][KDIM] bf16 bits
    const float* __restrict__ bias,         // [N]
    float* __restrict__ Cf,                 // [MPAD][N] (if F32OUT)
    unsigned short* __restrict__ Cb,        // [MPAD][N] (if BF16OUT)
    int N) {
  __shared__ unsigned short As[64][40];  // 40-elem stride: bank spread
  __shared__ unsigned short Bs[64][40];
  const int tid  = threadIdx.x;
  const int wave = tid >> 5;
  const int lane = tid & 31;
  const int m0 = blockIdx.x * 64;
  const int n0 = blockIdx.y * 64;
  const int wm = (wave & 1) * 32;
  const int wn = (wave >> 1) * 32;
  const int fl = lane & 15;   // A row / B col within 16
  const int kh = lane >> 4;   // lane-half selects K sub-block

#if USE_TDM
  const unsigned ldsA = (unsigned)(unsigned long long)(void*)&As[0][0];
  const unsigned ldsB = (unsigned)(unsigned long long)(void*)&Bs[0][0];
#else
  const int ldr = tid >> 1;         // 0..63 tile row to fetch
  const int ldc = (tid & 1) * 16;   // 0 or 16 (elems)
#endif

  v8f acc[2][2] = {};

  for (int k0 = 0; k0 < KDIM; k0 += 32) {
    __syncthreads();
#if USE_TDM
    if (tid == 0) {   // one wave issues both tile DMAs, waits, then barrier
      tdm_tile_load(A  + (size_t)m0 * KDIM + k0, ldsA, KDIM);
      tdm_tile_load(Bt + (size_t)n0 * KDIM + k0, ldsB, KDIM);
      __builtin_amdgcn_s_wait_tensorcnt(0);
    }
#else
    {
      const uint4* sa = (const uint4*)(A + (size_t)(m0 + ldr) * KDIM + k0 + ldc);
      *(uint4*)&As[ldr][ldc]     = sa[0];
      *(uint4*)&As[ldr][ldc + 8] = sa[1];
      const uint4* sb = (const uint4*)(Bt + (size_t)(n0 + ldr) * KDIM + k0 + ldc);
      *(uint4*)&Bs[ldr][ldc]     = sb[0];
      *(uint4*)&Bs[ldr][ldc + 8] = sb[1];
    }
#endif
    __syncthreads();

    union Frag { uint4 u[2]; v16bf v; } a[2], b[2];
#pragma unroll
    for (int i = 0; i < 2; ++i) {
      const int am = wm + i * 16 + fl;
      // A 16x32 bf16 layout: lane-half kh -> K = kh*8 + {0..7, 16..23}
      a[i].u[0] = *(const uint4*)&As[am][kh * 8];
      a[i].u[1] = *(const uint4*)&As[am][kh * 8 + 16];
      const int bn = wn + i * 16 + fl;
      // B 32x16 bf16 layout: lane-half kh -> K = kh*8 + {0..7, 16..23}
      b[i].u[0] = *(const uint4*)&Bs[bn][kh * 8];
      b[i].u[1] = *(const uint4*)&Bs[bn][kh * 8 + 16];
    }
#pragma unroll
    for (int mi = 0; mi < 2; ++mi)
#pragma unroll
      for (int ni = 0; ni < 2; ++ni)
        acc[mi][ni] = __builtin_amdgcn_wmma_f32_16x16x32_bf16(
            false, a[mi].v, false, b[ni].v, (short)0, acc[mi][ni], false, false);
  }

  // epilogue: C/D layout rows = 8*kh + r, col = fl
#pragma unroll
  for (int mi = 0; mi < 2; ++mi) {
    const int mbase = m0 + wm + mi * 16 + kh * 8;
#pragma unroll
    for (int ni = 0; ni < 2; ++ni) {
      const int n = n0 + wn + ni * 16 + fl;
      const float bv = bias ? bias[n] : 0.f;
#pragma unroll
      for (int r = 0; r < 8; ++r) {
        float v = acc[mi][ni][r] + bv;
        if (RELU) v = fmaxf(v, 0.f);
        const size_t o = (size_t)(mbase + r) * (unsigned)N + n;
        if (F32OUT) Cf[o] = v;
        if (BF16OUT) Cb[o] = f2bf(v);
      }
    }
  }
}

// ---------------- softmax over NL*NP=16 per (row, head) ----------------------
__global__ __launch_bounds__(256) void softmax16(float* __restrict__ attn) {
  int idx = blockIdx.x * 256 + threadIdx.x;
  if (idx >= M_N * NH_N) return;
  float* p = attn + (size_t)(idx >> 3) * 128 + (size_t)(idx & 7) * 16;
  float v[16], mx = -3.4e38f;
#pragma unroll
  for (int i = 0; i < 16; ++i) { v[i] = p[i]; mx = fmaxf(mx, v[i]); }
  float sum = 0.f;
#pragma unroll
  for (int i = 0; i < 16; ++i) { v[i] = __expf(v[i] - mx); sum += v[i]; }
  const float inv = 1.f / sum;
#pragma unroll
  for (int i = 0; i < 16; ++i) p[i] = v[i] * inv;
}

// ---------------- deformable trilinear sampling ------------------------------
// one wave per (row, head); lane = channel within head (HD=32)
__global__ __launch_bounds__(256) void deform_sample(
    const float* __restrict__ value,  // [MPAD][256]
    const float* __restrict__ off,    // [MPAD][384]
    const float* __restrict__ attn,   // [MPAD][128] (softmaxed)
    float* __restrict__ out) {        // [MPAD][256]
  const int gw = blockIdx.x * 8 + (threadIdx.x >> 5);
  if (gw >= M_N * NH_N) return;
  const int lane = threadIdx.x & 31;
  const int row = gw >> 3;
  const int h   = gw & 7;
  const int b = row / S_N;
  const int s = row - b * S_N;
  // reference point of query s
  const int ql = (s >= c_lvl_start[3]) ? 3 : (s >= c_lvl_start[2]) ? 2
               : (s >= c_lvl_start[1]) ? 1 : 0;
  const int sp = s - c_lvl_start[ql];
  const int qW = c_lvl_W[ql], qH = c_lvl_H[ql], qD = c_lvl_D[ql];
  const int izr = sp / (qH * qW);
  const int rem = sp - izr * qH * qW;
  const int iyr = rem / qW;
  const int ixr = rem - iyr * qW;
  const float rx = (ixr + 0.5f) / qW;
  const float ry = (iyr + 0.5f) / qH;
  const float rz = (izr + 0.5f) / qD;
  const int c = h * HD_N + lane;
  float accv = 0.f;
#pragma unroll
  for (int l = 0; l < NL_N; ++l) {
    const int D = c_lvl_D[l], H = c_lvl_H[l], W = c_lvl_W[l];
    const float* vbase = value + (size_t)(b * S_N + c_lvl_start[l]) * E_N + c;
#pragma unroll
    for (int p = 0; p < NP_N; ++p) {
      const size_t ob = (size_t)row * 384 + (size_t)(((h * NL_N + l) * NP_N + p) * 3);
      const float ox = off[ob], oy = off[ob + 1], oz = off[ob + 2];
      const float aw = attn[(size_t)row * 128 + (h * NL_N + l) * NP_N + p];
      // loc = ref + off/[W,H,D];  coord = loc*[W,H,D] - 0.5
      const float cx = rx * W + ox - 0.5f;
      const float cy = ry * H + oy - 0.5f;
      const float cz = rz * D + oz - 0.5f;
      const float fx0 = floorf(cx), fy0 = floorf(cy), fz0 = floorf(cz);
      const float fx = cx - fx0, fy = cy - fy0, fz = cz - fz0;
      const int x0 = (int)fx0, y0 = (int)fy0, z0 = (int)fz0;
#pragma unroll
      for (int dz = 0; dz < 2; ++dz) {
        const int z = z0 + dz;
        const float wz = dz ? fz : 1.f - fz;
        const bool okz = (z >= 0) && (z < D);
        const int zc = min(max(z, 0), D - 1);
#pragma unroll
        for (int dy = 0; dy < 2; ++dy) {
          const int y = y0 + dy;
          const float wy = dy ? fy : 1.f - fy;
          const bool oky = (y >= 0) && (y < H);
          const int yc = min(max(y, 0), H - 1);
#pragma unroll
          for (int dx = 0; dx < 2; ++dx) {
            const int x = x0 + dx;
            const float wx = dx ? fx : 1.f - fx;
            const bool okx = (x >= 0) && (x < W);
            const int xc = min(max(x, 0), W - 1);
            const float wgt = (okx && oky && okz) ? aw * wx * wy * wz : 0.f;
            const float g = vbase[(size_t)((zc * H + yc) * W + xc) * E_N];
            accv += wgt * g;
          }
        }
      }
    }
  }
  out[(size_t)row * E_N + c] = accv;
}

// ---------------- residual + LayerNorm (wave per row, lane-strided) ----------
__global__ __launch_bounds__(256) void residual_layernorm(
    const float* __restrict__ x, const float* __restrict__ d,
    const float* __restrict__ gamma, const float* __restrict__ beta,
    float* __restrict__ out) {
  const int rowi = blockIdx.x * 8 + (threadIdx.x >> 5);
  if (rowi >= M_N) return;
  const int lane = threadIdx.x & 31;
  const size_t base = (size_t)rowi * E_N;
  float v[8];
  float sum = 0.f;
#pragma unroll
  for (int i = 0; i < 8; ++i) {
    const int e = lane + i * 32;
    v[i] = x[base + e] + d[base + e];
    sum += v[i];
  }
#pragma unroll
  for (int o = 16; o > 0; o >>= 1) sum += __shfl_xor(sum, o, 32);
  const float mean = sum * (1.f / 256.f);
  float var = 0.f;
#pragma unroll
  for (int i = 0; i < 8; ++i) { const float t = v[i] - mean; var += t * t; }
#pragma unroll
  for (int o = 16; o > 0; o >>= 1) var += __shfl_xor(var, o, 32);
  const float rstd = rsqrtf(var * (1.f / 256.f) + 1e-5f);
#pragma unroll
  for (int i = 0; i < 8; ++i) {
    const int e = lane + i * 32;
    out[base + e] = (v[i] - mean) * rstd * gamma[e] + beta[e];
  }
}

// ============================ host launcher ==================================
extern "C" void kernel_launch(void* const* d_in, const int* in_sizes, int n_in,
                              void* d_out, int out_size, void* d_ws, size_t ws_size,
                              hipStream_t stream) {
  (void)in_sizes; (void)n_in; (void)out_size; (void)ws_size;
  const float* feat[4] = {(const float*)d_in[0], (const float*)d_in[1],
                          (const float*)d_in[2], (const float*)d_in[3]};
  const float* posi[4] = {(const float*)d_in[4], (const float*)d_in[5],
                          (const float*)d_in[6], (const float*)d_in[7]};
  const float* lvl_emb = (const float*)d_in[8];
  const float* W_off  = (const float*)d_in[9];  const float* b_off  = (const float*)d_in[10];
  const float* W_attn = (const float*)d_in[11]; const float* b_attn = (const float*)d_in[12];
  const float* W_val  = (const float*)d_in[13]; const float* b_val  = (const float*)d_in[14];
  const float* W_out  = (const float*)d_in[15]; const float* b_out  = (const float*)d_in[16];
  const float* ln1_g  = (const float*)d_in[17]; const float* ln1_b  = (const float*)d_in[18];
  const float* W_ff1  = (const float*)d_in[19]; const float* b_ff1  = (const float*)d_in[20];
  const float* W_ff2  = (const float*)d_in[21]; const float* b_ff2  = (const float*)d_in[22];
  const float* ln2_g  = (const float*)d_in[23]; const float* ln2_b  = (const float*)d_in[24];

  size_t cur = 0;
  auto carve = [&](size_t bytes) -> void* {
    void* p = (char*)d_ws + cur;
    cur += (bytes + 255) & ~(size_t)255;
    return p;
  };
  float* xf    = (float*)carve((size_t)MPAD_N * 256 * 4);
  float* posf  = (float*)carve((size_t)MPAD_N * 256 * 4);
  unsigned short* xbf    = (unsigned short*)carve((size_t)MPAD_N * 256 * 2);
  unsigned short* qbf    = (unsigned short*)carve((size_t)MPAD_N * 256 * 2);
  float* valuef = (float*)carve((size_t)MPAD_N * 256 * 4);
  float* offf   = (float*)carve((size_t)MPAD_N * 384 * 4);
  float* attnf  = (float*)carve((size_t)MPAD_N * 128 * 4);
  float* sampf  = (float*)carve((size_t)MPAD_N * 256 * 4);
  unsigned short* sampbf = (unsigned short*)carve((size_t)MPAD_N * 256 * 2);
  float* projf  = (float*)carve((size_t)MPAD_N * 256 * 4);
  float* x1f    = (float*)carve((size_t)MPAD_N * 256 * 4);
  unsigned short* x1bf   = (unsigned short*)carve((size_t)MPAD_N * 256 * 2);
  unsigned short* hidbf  = (unsigned short*)carve((size_t)MPAD_N * 1024 * 2);
  float* ff2f   = (float*)carve((size_t)MPAD_N * 256 * 4);
  unsigned short* wofft  = (unsigned short*)carve((size_t)NLAYERS_N * 384 * 256 * 2);
  unsigned short* wattnt = (unsigned short*)carve((size_t)NLAYERS_N * 128 * 256 * 2);
  unsigned short* wvalt  = (unsigned short*)carve((size_t)NLAYERS_N * 256 * 256 * 2);
  unsigned short* woutt  = (unsigned short*)carve((size_t)NLAYERS_N * 256 * 256 * 2);
  unsigned short* wff1t  = (unsigned short*)carve((size_t)NLAYERS_N * 1024 * 256 * 2);
  unsigned short* wff2t  = (unsigned short*)carve((size_t)NLAYERS_N * 256 * 1024 * 2);

  // ---- pack inputs ----
  pack_x_pos<<<MPAD_N, 256, 0, stream>>>(feat[0], feat[1], feat[2], feat[3],
                                         posi[0], posi[1], posi[2], posi[3],
                                         lvl_emb, xf, posf);
  // ---- one-time weight transpose+convert ----
  for (int l = 0; l < NLAYERS_N; ++l) {
    wt_transpose<<<(384 * 256) / 256, 256, 0, stream>>>(
        W_off + (size_t)l * 256 * 384, wofft + (size_t)l * 384 * 256, 256, 384);
    wt_transpose<<<(128 * 256) / 256, 256, 0, stream>>>(
        W_attn + (size_t)l * 256 * 128, wattnt + (size_t)l * 128 * 256, 256, 128);
    wt_transpose<<<(256 * 256) / 256, 256, 0, stream>>>(
        W_val + (size_t)l * 256 * 256, wvalt + (size_t)l * 256 * 256, 256, 256);
    wt_transpose<<<(256 * 256) / 256, 256, 0, stream>>>(
        W_out + (size_t)l * 256 * 256, woutt + (size_t)l * 256 * 256, 256, 256);
    wt_transpose<<<(1024 * 256) / 256, 256, 0, stream>>>(
        W_ff1 + (size_t)l * 256 * 1024, wff1t + (size_t)l * 1024 * 256, 256, 1024);
    wt_transpose<<<(256 * 1024) / 256, 256, 0, stream>>>(
        W_ff2 + (size_t)l * 1024 * 256, wff2t + (size_t)l * 256 * 1024, 1024, 256);
  }

  const dim3 gblk(128);
  const int gM = MPAD_N / 64;  // 330
  float* xcur = xf;

  for (int l = 0; l < NLAYERS_N; ++l) {
    // activations -> bf16
    cvt_bf16_rows<<<MPAD_N, 256, 0, stream>>>(xcur, xbf, 256);
    add_cvt_bf16_rows<<<MPAD_N, 256, 0, stream>>>(xcur, posf, qbf);

    // value = x @ W_val + b_val
    gemm_bf16_wmma<256, false, true, false><<<dim3(gM, 256 / 64), gblk, 0, stream>>>(
        xbf, wvalt + (size_t)l * 256 * 256, b_val + (size_t)l * 256, valuef, nullptr, 256);
    // off = q @ W_off + b_off
    gemm_bf16_wmma<256, false, true, false><<<dim3(gM, 384 / 64), gblk, 0, stream>>>(
        qbf, wofft + (size_t)l * 384 * 256, b_off + (size_t)l * 384, offf, nullptr, 384);
    // attn_logits = q @ W_attn + b_attn
    gemm_bf16_wmma<256, false, true, false><<<dim3(gM, 128 / 64), gblk, 0, stream>>>(
        qbf, wattnt + (size_t)l * 128 * 256, b_attn + (size_t)l * 128, attnf, nullptr, 128);

    softmax16<<<(M_N * NH_N + 255) / 256, 256, 0, stream>>>(attnf);
    deform_sample<<<(M_N * NH_N + 7) / 8, 256, 0, stream>>>(valuef, offf, attnf, sampf);

    cvt_bf16_rows<<<MPAD_N, 256, 0, stream>>>(sampf, sampbf, 256);
    // proj = samp @ W_out + b_out
    gemm_bf16_wmma<256, false, true, false><<<dim3(gM, 256 / 64), gblk, 0, stream>>>(
        sampbf, woutt + (size_t)l * 256 * 256, b_out + (size_t)l * 256, projf, nullptr, 256);

    residual_layernorm<<<(M_N + 7) / 8, 256, 0, stream>>>(
        xcur, projf, ln1_g + (size_t)l * 256, ln1_b + (size_t)l * 256, x1f);

    cvt_bf16_rows<<<MPAD_N, 256, 0, stream>>>(x1f, x1bf, 256);
    // hidden = relu(x1 @ W_ff1 + b_ff1) -> bf16 direct
    gemm_bf16_wmma<256, true, false, true><<<dim3(gM, 1024 / 64), gblk, 0, stream>>>(
        x1bf, wff1t + (size_t)l * 1024 * 256, b_ff1 + (size_t)l * 1024, nullptr, hidbf, 1024);
    // ff2 = hidden @ W_ff2 + b_ff2
    gemm_bf16_wmma<1024, false, true, false><<<dim3(gM, 256 / 64), gblk, 0, stream>>>(
        hidbf, wff2t + (size_t)l * 256 * 1024, b_ff2 + (size_t)l * 256, ff2f, nullptr, 256);

    float* xnext = (l == NLAYERS_N - 1) ? (float*)d_out : xf;
    residual_layernorm<<<(M_N + 7) / 8, 256, 0, stream>>>(
        x1f, ff2f, ln2_g + (size_t)l * 256, ln2_b + (size_t)l * 256, xnext);
    xcur = xnext;
  }
}
